// WindowAttention_81518479278502
// MI455X (gfx1250) — matrix operand order repaired and checked by reading, compiled, not run on hardware
//
#include <hip/hip_runtime.h>
#include <math.h>

typedef __attribute__((ext_vector_type(16))) _Float16 v16h;
typedef __attribute__((ext_vector_type(8)))  float    v8f;

#define SEQ    2048
#define NH     8
#define DH     64
#define HW     64          // half window (WINDOW-1)/2
#define NCHUNK 5           // ceil((16 + 2*HW)/32) chunks of 32 keys
#define WPB    4           // waves per block

__device__ __forceinline__ int clamp_seq(int i) {
    return i < 0 ? 0 : (i > SEQ - 1 ? SEQ - 1 : i);
}

__global__ __launch_bounds__(32 * WPB)
void swin_attn_wmma_kernel(const float* __restrict__ qg,
                           const float* __restrict__ kg,
                           const float* __restrict__ vg,
                           float* __restrict__ outg)
{
    const int lane = threadIdx.x & 31;
    const int wave = threadIdx.x >> 5;
    const int hd   = blockIdx.y;                       // head
    const int q0   = (blockIdx.x * WPB + wave) * 16;   // first query of this wave
    const int ln   = lane & 15;                        // N column (query) / A row
    const int hi   = lane >> 4;                        // half-wave id
    const int RS   = NH * DH;                          // 512 floats per seq step

    // ---- B fragments of Q^T (32x16 per k-step): lane -> query q0+ln,
    //      VGPR p -> dim = ks*32 + 2*(p&3) + 16*(p>>2) + 8*hi.  Pre-scaled by d^-0.5 = 1/8.
    v16h bq[2];
    {
        const float* qrow = qg + (size_t)(q0 + ln) * RS + hd * DH;
#pragma unroll
        for (int ks = 0; ks < 2; ++ks)
#pragma unroll
            for (int p = 0; p < 8; ++p) {
                int d = ks * 32 + ((p & 3) << 1) + ((p >> 2) << 4) + (hi << 3);
                bq[ks][2 * p]     = (_Float16)(qrow[d]     * 0.125f);
                bq[ks][2 * p + 1] = (_Float16)(qrow[d + 1] * 0.125f);
            }
    }

    // online-softmax state: one query per lane (query = q0 + ln) -> all
    // rescale/normalize state is a per-lane scalar.  O^T accumulators:
    // element r of oacc[nt] <-> dim nt*16 + r + 8*hi, lane <-> query.
    float mrun = -1.0e30f, lrun = 0.0f;
    v8f oacc[4];
#pragma unroll
    for (int nt = 0; nt < 4; ++nt)
#pragma unroll
        for (int r = 0; r < 8; ++r) oacc[nt][r] = 0.0f;

    for (int c = 0; c < NCHUNK; ++c) {
        const int kBase = q0 - HW + c * 32;            // first key of this chunk

        // prefetch next chunk's K and V rows; clamped -> no divergent branch
        {
            int kpf = clamp_seq(kBase + 32 + lane);
            const float* kp = kg + (size_t)kpf * RS + hd * DH;
            const float* vp = vg + (size_t)kpf * RS + hd * DH;
            __builtin_prefetch(kp, 0, 1);
            __builtin_prefetch(kp + 32, 0, 1);
            __builtin_prefetch(vp, 0, 1);
            __builtin_prefetch(vp + 32, 0, 1);
        }

        // ---- S^T = K * Q^T : two 16-key tiles, two K=32 steps over d each.
        //      C layout: element r <-> key kBase + 16T + r + 8*hi, lane <-> query.
        //      OOB key rows are CLAMPED (their scores get masked to -inf below).
        v8f st[2];
#pragma unroll
        for (int T = 0; T < 2; ++T) {
#pragma unroll
            for (int r = 0; r < 8; ++r) st[T][r] = 0.0f;
            const int krow  = clamp_seq(kBase + 16 * T + ln);   // A row = key
            const float* kp = kg + (size_t)krow * RS + hd * DH;
#pragma unroll
            for (int ks = 0; ks < 2; ++ks) {
                v16h ak;
#pragma unroll
                for (int p = 0; p < 8; ++p) {
                    int kb = ((p & 3) << 1) + ((p >> 2) << 4) + (hi << 3);
                    int d  = ks * 32 + kb;
                    ak[2 * p]     = (_Float16)kp[d];
                    ak[2 * p + 1] = (_Float16)kp[d + 1];
                }
                st[T] = __builtin_amdgcn_wmma_f32_16x16x32_f16(false, ak, false, bq[ks],
                                                               (short)0, st[T],
                                                               false, false);
            }
        }

        // ---- mask + online softmax.  Each lane holds 16 of the chunk's 32 keys
        //      for its query; the half-wave partner (lane^16) holds the other 16.
        const int qi = q0 + ln;
        float sc[16];
#pragma unroll
        for (int T = 0; T < 2; ++T)
#pragma unroll
            for (int r = 0; r < 8; ++r) {
                int ki = kBase + 16 * T + r + 8 * hi;
                bool valid = ((unsigned)ki < (unsigned)SEQ) &&
                             ((unsigned)(ki - qi + HW) <= (unsigned)(2 * HW));
                sc[T * 8 + r] = valid ? st[T][r] : -__builtin_inff();
            }
        float mx = sc[0];
#pragma unroll
        for (int j = 1; j < 16; ++j) mx = fmaxf(mx, sc[j]);
        mx = fmaxf(mx, __shfl_xor(mx, 16, 32));        // combine half-waves
        const float mnew = fmaxf(mrun, mx);
        const float corr = __expf(mrun - mnew);

        float ps[16];
        float sum = 0.0f;
#pragma unroll
        for (int j = 0; j < 16; ++j) {
            ps[j] = __expf(sc[j] - mnew);              // masked -> exp(-inf) = 0
            sum  += ps[j];
        }
        sum += __shfl_xor(sum, 16, 32);
        lrun = lrun * corr + sum;
        mrun = mnew;
#pragma unroll
        for (int nt = 0; nt < 4; ++nt)
#pragma unroll
            for (int r = 0; r < 8; ++r) oacc[nt][r] *= corr;

        // ---- build P^T B-fragment (32x16): lane (ln,hi) needs key offsets
        //      16*hi + j (j=0..15) of query ln.  Own lane supplies j = 8*hi + r
        //      (tile T = hi); partner lane supplies the other 8 via one xor-16.
        float a_self[8], a_part[8], a_oth[8];
#pragma unroll
        for (int r = 0; r < 8; ++r) {
            a_self[r] = hi ? ps[8 + r] : ps[r];        // tile T = own hi
            a_part[r] = hi ? ps[r]     : ps[8 + r];    // tile T = partner's hi
        }
#pragma unroll
        for (int r = 0; r < 8; ++r)
            a_oth[r] = __shfl_xor(a_part[r], 16, 32);

        v16h bp;
#pragma unroll
        for (int j = 0; j < 16; ++j) {
            float pj = ((j >> 3) == hi) ? a_self[j & 7] : a_oth[j & 7];
            bp[j] = (_Float16)pj;
        }

        // ---- O^T += V^T * P^T : 4 dim-tiles, K = 32 real keys (no padding).
        //      OOB key rows are CLAMPED: their P weight is exactly 0.
#pragma unroll
        for (int nt = 0; nt < 4; ++nt) {
            v16h av;                                   // A row = dim nt*16 + ln
#pragma unroll
            for (int p = 0; p < 8; ++p) {
                int kb  = (hi << 4) + 2 * p;
                int kn0 = clamp_seq(kBase + kb);
                int kn1 = clamp_seq(kBase + kb + 1);
                size_t col = (size_t)(hd * DH + nt * 16 + ln);
                av[2 * p]     = (_Float16)vg[(size_t)kn0 * RS + col];
                av[2 * p + 1] = (_Float16)vg[(size_t)kn1 * RS + col];
            }
            oacc[nt] = __builtin_amdgcn_wmma_f32_16x16x32_f16(false, av, false, bp,
                                                              (short)0, oacc[nt],
                                                              false, false);
        }
    }

    // ---- normalize and store.  O^T element r <-> dim nt*16 + 8*hi + r,
    //      lane <-> query: 8 consecutive floats per (lane,nt) -> wide stores.
    const float linv = 1.0f / lrun;
    float* orow = outg + (size_t)(q0 + ln) * RS + hd * DH + 8 * hi;
#pragma unroll
    for (int nt = 0; nt < 4; ++nt)
#pragma unroll
        for (int r = 0; r < 8; ++r)
            orow[nt * 16 + r] = oacc[nt][r] * linv;
}

extern "C" void kernel_launch(void* const* d_in, const int* in_sizes, int n_in,
                              void* d_out, int out_size, void* d_ws, size_t ws_size,
                              hipStream_t stream)
{
    (void)in_sizes; (void)n_in; (void)out_size; (void)d_ws; (void)ws_size;
    const float* q = (const float*)d_in[0];
    const float* k = (const float*)d_in[1];
    const float* v = (const float*)d_in[2];
    float* out = (float*)d_out;

    dim3 grid(SEQ / (16 * WPB), NH, 1);   // 32 x 8 blocks
    dim3 block(32 * WPB, 1, 1);           // 4 waves of 32
    hipLaunchKernelGGL(swin_attn_wmma_kernel, grid, block, 0, stream, q, k, v, out);
}
